// BertSelfAttention_42752104465209
// MI455X (gfx1250) — compile-verified, matrix operand order checked by reading, not run on hardware
//
#include <hip/hip_runtime.h>

typedef _Float16 v16h __attribute__((ext_vector_type(16)));
typedef float    v8f  __attribute__((ext_vector_type(8)));

// Two 16-byte chunks -> one 16-halfword WMMA fragment (A or B operand).
// Fragment element j corresponds to K index (j<8?0:16) + 8*half + (j%8),
// so chunk0 sits at +8*half and chunk1 at +16+8*half within a 32-wide K slab.
struct FragBits { uint4 lo, hi; };

static __device__ __forceinline__ v16h load_frag(const _Float16* slab, int half) {
  FragBits u;
  u.lo = *reinterpret_cast<const uint4*>(slab + 8 * half);
  u.hi = *reinterpret_cast<const uint4*>(slab + 16 + 8 * half);
  return __builtin_bit_cast(v16h, u);
}

// ---------------- fp32 -> f16 convert ----------------
__global__ void cvt_f32_f16_kernel(const float* __restrict__ x,
                                   _Float16* __restrict__ y, int n) {
  int i = blockIdx.x * blockDim.x + threadIdx.x;
  int stride = gridDim.x * blockDim.x;
  for (; i < n; i += stride) y[i] = (_Float16)x[i];
}

// ---------------- W [K,N] fp32 -> Wt [N,K] f16 (tiled transpose) ----------------
__global__ void transpose_w_kernel(const float* __restrict__ W,
                                   _Float16* __restrict__ Wt) {
  __shared__ float tile[32][33];
  const int bx = blockIdx.x * 32;  // n dimension
  const int by = blockIdx.y * 32;  // k dimension
  const int tx = threadIdx.x, ty = threadIdx.y;
#pragma unroll
  for (int j = 0; j < 32; j += 8)
    tile[ty + j][tx] = W[(by + ty + j) * 1024 + bx + tx];
  __syncthreads();
#pragma unroll
  for (int j = 0; j < 32; j += 8)
    Wt[(bx + ty + j) * 1024 + by + tx] = (_Float16)tile[tx][ty + j];
}

// ---------------- QKV projection: out = X @ W + b, f16 WMMA, f32 accum ----------
// One wave computes a 16x64 tile. v_layout: 0 -> [b,h,S,d] (Q,K), 1 -> [b,h,d,S] (V^T)
__global__ void qkv_gemm_kernel(const _Float16* __restrict__ Xh,
                                const _Float16* __restrict__ Wt,
                                const float* __restrict__ bias,
                                _Float16* __restrict__ out, int v_layout) {
  const int lane = threadIdx.x & 31;
  const int wv = threadIdx.x >> 5;
  const int wid = blockIdx.x * 4 + wv;      // 4096 waves
  const int r = wid & 255;                  // row tile (16 rows of 4096)
  const int c = wid >> 8;                   // col block (64 cols of 1024)
  const int ln = lane & 15, half = lane >> 4;

  v8f acc[4];
#pragma unroll
  for (int t = 0; t < 4; ++t) acc[t] = {};

  const _Float16* arow = Xh + (r * 16 + ln) * 1024;
  for (int kk = 0; kk < 32; ++kk) {
    v16h a = load_frag(arow + kk * 32, half);
#pragma unroll
    for (int t = 0; t < 4; ++t) {
      const _Float16* brow = Wt + (c * 64 + t * 16 + ln) * 1024 + kk * 32;
      v16h b = load_frag(brow, half);
      acc[t] = __builtin_amdgcn_wmma_f32_16x16x32_f16(false, a, false, b,
                                                      (short)0, acc[t], false, false);
    }
  }

#pragma unroll
  for (int t = 0; t < 4; ++t) {
    const int col = c * 64 + t * 16 + ln;   // 0..1023
    const float bz = bias[col];
    const int hh = col >> 6, dd = col & 63;
#pragma unroll
    for (int i = 0; i < 8; ++i) {
      const int row = r * 16 + i + 8 * half;  // 0..4095
      const int bb = row >> 11, ss = row & 2047;
      const float val = acc[t][i] + bz;
      if (v_layout == 0)
        out[((bb * 16 + hh) * 2048 + ss) * 64 + dd] = (_Float16)val;
      else
        out[((bb * 16 + hh) * 64 + dd) * 2048 + ss] = (_Float16)val;
    }
  }
}

// ---------------- fused flash attention ----------------
// One wave per 16-query tile per (b,h). Streams 32-key blocks, online softmax.
__global__ void attn_kernel(const _Float16* __restrict__ Qh,
                            const _Float16* __restrict__ Kh,
                            const _Float16* __restrict__ Vth,
                            const float* __restrict__ mask,
                            float* __restrict__ out) {
  __shared__ _Float16 plds[4][16 * 40];   // P tile per wave, pitch 40 halves (16B aligned rows)
  const int lane = threadIdx.x & 31;
  const int wv = threadIdx.x >> 5;
  const int g = blockIdx.x * 4 + wv;      // 4096 waves
  const int qt = g & 127;                 // query tile 0..127
  const int bh = g >> 7;                  // 0..31
  const int b = bh >> 4, h = bh & 15;
  const int ln = lane & 15, half = lane >> 4;
  const int qbase = qt * 16;

  const _Float16* qrow = Qh + (bh * 2048 + qbase + ln) * 64;
  v16h aq0 = load_frag(qrow, half);
  v16h aq1 = load_frag(qrow + 32, half);

  float ms[8], ls[8];
  v8f acc[4];
#pragma unroll
  for (int i = 0; i < 8; ++i) { ms[i] = -1e30f; ls[i] = 0.f; }
#pragma unroll
  for (int t = 0; t < 4; ++t) acc[t] = {};

  _Float16* pr = &plds[wv][0];

  for (int kb = 0; kb < 64; ++kb) {
    const int kbase = kb * 32;
    v8f s[2];
#pragma unroll
    for (int t = 0; t < 2; ++t) {
      const _Float16* krow = Kh + (bh * 2048 + kbase + t * 16 + ln) * 64;
      v16h bk0 = load_frag(krow, half);
      v16h bk1 = load_frag(krow + 32, half);
      v8f cc = {};
      cc = __builtin_amdgcn_wmma_f32_16x16x32_f16(false, aq0, false, bk0, (short)0, cc, false, false);
      cc = __builtin_amdgcn_wmma_f32_16x16x32_f16(false, aq1, false, bk1, (short)0, cc, false, false);
      const float mv = mask[b * 2048 + kbase + t * 16 + ln];
#pragma unroll
      for (int i = 0; i < 8; ++i) cc[i] = cc[i] * 0.125f + mv;
      s[t] = cc;
    }

    // online softmax: each row (m = i + 8*half) lives across one 16-lane group
    float mn[8], scale[8];
#pragma unroll
    for (int i = 0; i < 8; ++i) {
      float t0 = fmaxf(s[0][i], s[1][i]);
#pragma unroll
      for (int off = 1; off < 16; off <<= 1)
        t0 = fmaxf(t0, __shfl_xor(t0, off, 32));
      mn[i] = fmaxf(ms[i], t0);
      scale[i] = __expf(ms[i] - mn[i]);
      ms[i] = mn[i];
    }
#pragma unroll
    for (int i = 0; i < 8; ++i) {
      const float p0 = __expf(s[0][i] - mn[i]);
      const float p1 = __expf(s[1][i] - mn[i]);
      s[0][i] = p0; s[1][i] = p1;
      float t1 = p0 + p1;
#pragma unroll
      for (int off = 1; off < 16; off <<= 1)
        t1 += __shfl_xor(t1, off, 32);
      ls[i] = ls[i] * scale[i] + t1;
    }

    // transpose P (D-fragment layout -> A-fragment layout) via LDS
#pragma unroll
    for (int i = 0; i < 8; ++i) {
      const int row = i + 8 * half;
      pr[row * 40 + ln]      = (_Float16)s[0][i];
      pr[row * 40 + 16 + ln] = (_Float16)s[1][i];
    }
    asm volatile("s_wait_dscnt 0" ::: "memory");
    v16h ap = load_frag(pr + ln * 40, half);

#pragma unroll
    for (int td = 0; td < 4; ++td) {
#pragma unroll
      for (int i = 0; i < 8; ++i) acc[td][i] *= scale[i];
      const _Float16* vrow = Vth + (bh * 64 + td * 16 + ln) * 2048 + kbase;
      v16h bvf = load_frag(vrow, half);
      acc[td] = __builtin_amdgcn_wmma_f32_16x16x32_f16(false, ap, false, bvf,
                                                       (short)0, acc[td], false, false);
    }
  }

#pragma unroll
  for (int i = 0; i < 8; ++i) {
    const float inv = 1.0f / ls[i];
    const int srow = qbase + i + 8 * half;
#pragma unroll
    for (int td = 0; td < 4; ++td)
      out[(b * 2048 + srow) * 1024 + h * 64 + td * 16 + ln] = acc[td][i] * inv;
  }
}

extern "C" void kernel_launch(void* const* d_in, const int* in_sizes, int n_in,
                              void* d_out, int out_size, void* d_ws, size_t ws_size,
                              hipStream_t stream) {
  const float* X    = (const float*)d_in[0];
  const float* mask = (const float*)d_in[1];
  const float* Wq   = (const float*)d_in[2];
  const float* bq   = (const float*)d_in[3];
  const float* Wk   = (const float*)d_in[4];
  const float* bk   = (const float*)d_in[5];
  const float* Wv   = (const float*)d_in[6];
  const float* bv   = (const float*)d_in[7];
  float* out = (float*)d_out;

  _Float16* ws  = (_Float16*)d_ws;
  _Float16* Xh  = ws;                    // 4096*1024
  _Float16* Wtq = Xh + 4194304;          // 1024*1024 each
  _Float16* Wtk = Wtq + 1048576;
  _Float16* Wtv = Wtk + 1048576;
  _Float16* Qh  = Wtv + 1048576;         // [2,16,2048,64]
  _Float16* Kh  = Qh + 4194304;
  _Float16* Vth = Kh + 4194304;          // [2,16,64,2048]

  cvt_f32_f16_kernel<<<2048, 256, 0, stream>>>(X, Xh, 4194304);
  dim3 tb(32, 8), tg(32, 32);
  transpose_w_kernel<<<tg, tb, 0, stream>>>(Wq, Wtq);
  transpose_w_kernel<<<tg, tb, 0, stream>>>(Wk, Wtk);
  transpose_w_kernel<<<tg, tb, 0, stream>>>(Wv, Wtv);

  qkv_gemm_kernel<<<1024, 128, 0, stream>>>(Xh, Wtq, bq, Qh, 0);
  qkv_gemm_kernel<<<1024, 128, 0, stream>>>(Xh, Wtk, bk, Kh, 0);
  qkv_gemm_kernel<<<1024, 128, 0, stream>>>(Xh, Wtv, bv, Vth, 1);

  attn_kernel<<<1024, 128, 0, stream>>>(Qh, Kh, Vth, mask, out);
}
